// SoccerViT_14645838479542
// MI455X (gfx1250) — matrix operand order repaired and checked by reading, not compiled, not run on hardware
//
#include <hip/hip_runtime.h>
#include <math.h>

#define LNUM 8
#define DHID 768
#define NH   12
#define DHEAD 64
#define MLPD 3072
#define BB   32
#define SS   577
#define TT   (BB * SS)   // 18464 tokens

typedef _Float16 half_t;
typedef __attribute__((ext_vector_type(16))) _Float16 v16h;
typedef __attribute__((ext_vector_type(8)))  float    v8f;

// ---------------------------------------------------------------------------
// CDNA5 async global->LDS staging via inline asm (ASYNCcnt-tracked).
// ISA: GLOBAL_LOAD_ASYNC_TO_LDS_B128 — VDST = LDS byte address VGPR,
// VADDR = 64-bit global address, saddr NULL ("off") mode.
// LDS offset = low 32 bits of the generic LDS pointer (flat LDS aperture).
// ---------------------------------------------------------------------------
__device__ __forceinline__ void async_copy_b128(const half_t* g, half_t* l) {
#if defined(__AMDGCN__)
  unsigned loff = (unsigned)(unsigned long long)(const void*)l;
  asm volatile("global_load_async_to_lds_b128 %0, %1, off"
               :
               : "v"(loff), "v"((unsigned long long)(const void*)g)
               : "memory");
#else
  const unsigned long long* s = (const unsigned long long*)g;
  unsigned long long* d = (unsigned long long*)l;
  d[0] = s[0];
  d[1] = s[1];
#endif
}

__device__ __forceinline__ void async_wait0() {
#if defined(__AMDGCN__)
  asm volatile("s_wait_asynccnt 0x0" ::: "memory");
#endif
}

// copy 32 f16 (64B) from global to LDS (async)
__device__ __forceinline__ void stage32h(const half_t* src, half_t* dst) {
  async_copy_b128(src, dst);
  async_copy_b128(src + 8, dst + 8);
  async_copy_b128(src + 16, dst + 16);
  async_copy_b128(src + 24, dst + 24);
}
// copy 16 f16 (32B) from global to LDS (async)
__device__ __forceinline__ void stage16h(const half_t* src, half_t* dst) {
  async_copy_b128(src, dst);
  async_copy_b128(src + 8, dst + 8);
}

// ---------------------------------------------------------------------------
// WMMA helpers (gfx1250 wave32). Fragment layouts used here (A/B k order is this code's own, same for both):
//  A (16x32 f16): lane l row l%16; half i -> K=(l<16?0:16)+i
//  B (32x16 f16): lane l col l%16; half i -> K=(l<16?0:16)+i
//  C/D (16x16 f32): lane l col l%16; float r -> row r + (l<16?0:8)
// ---------------------------------------------------------------------------
__device__ __forceinline__ v8f wmma_f16(v16h a, v16h b, v8f c) {
  return __builtin_amdgcn_wmma_f32_16x16x32_f16(false, a, false, b, (short)0, c,
                                                false, false);
}

__device__ __forceinline__ v16h load_frag_a(const half_t* sm, int ldk, int r0,
                                            int k0, int lane) {
  const half_t* p = sm + (r0 + (lane & 15)) * ldk + k0 + ((lane < 16) ? 0 : 16);
  v16h o;
#pragma unroll
  for (int i = 0; i < 8; ++i) o[i] = p[i];
#pragma unroll
  for (int i = 0; i < 8; ++i) o[8 + i] = p[8 + i];
  return o;
}

__device__ __forceinline__ v16h load_frag_b(const half_t* sm, int ldk, int n0,
                                            int k0, int lane) {
  const half_t* p = sm + (n0 + (lane & 15)) * ldk + k0 + ((lane < 16) ? 0 : 16);
  v16h o;
#pragma unroll
  for (int i = 0; i < 16; ++i) o[i] = p[i];
  return o;
}

// ---------------------------------------------------------------------------
// residual init: d_out = x
// ---------------------------------------------------------------------------
__global__ void copy_f4_kernel(const float4* __restrict__ src,
                               float4* __restrict__ dst, int n) {
  int i = blockIdx.x * blockDim.x + threadIdx.x;
  if (i < n) dst[i] = src[i];
}

// ---------------------------------------------------------------------------
// LayerNorm over D=768, one 256-thread block per token, fp32 in -> f16 out
// ---------------------------------------------------------------------------
__global__ void __launch_bounds__(256)
ln_kernel(const float* __restrict__ x, const float* __restrict__ g,
          const float* __restrict__ b, half_t* __restrict__ out) {
  __shared__ float red[256];
  const int t = blockIdx.x, tid = threadIdx.x;
  const float* xp = x + (long long)t * DHID;
  float v0 = xp[tid], v1 = xp[tid + 256], v2 = xp[tid + 512];
  red[tid] = v0 + v1 + v2;
  __syncthreads();
  for (int o = 128; o > 0; o >>= 1) {
    if (tid < o) red[tid] += red[tid + o];
    __syncthreads();
  }
  float mean = red[0] * (1.0f / DHID);
  __syncthreads();
  float d0 = v0 - mean, d1 = v1 - mean, d2 = v2 - mean;
  red[tid] = d0 * d0 + d1 * d1 + d2 * d2;
  __syncthreads();
  for (int o = 128; o > 0; o >>= 1) {
    if (tid < o) red[tid] += red[tid + o];
    __syncthreads();
  }
  float rstd = rsqrtf(red[0] * (1.0f / DHID) + 1e-5f);
  half_t* op = out + (long long)t * DHID;
  op[tid]       = (half_t)(d0 * rstd * g[tid]       + b[tid]);
  op[tid + 256] = (half_t)(d1 * rstd * g[tid + 256] + b[tid + 256]);
  op[tid + 512] = (half_t)(d2 * rstd * g[tid + 512] + b[tid + 512]);
}

#define LDT 40

// ---------------------------------------------------------------------------
// Batched small GEMM (QKV):  C[M,N] = outScale*(A.W^T + bias)
// Block 128 thr / 4 waves, tile 64x64, K panel 32; f16 out.
// ---------------------------------------------------------------------------
__global__ void __launch_bounds__(128)
gemm64_kernel(const half_t* __restrict__ A, long long sAz1, long long sAz2,
              int zdiv, int lda, const float* __restrict__ W, long long sWz,
              int ldw, const float* __restrict__ bias, long long sBz,
              half_t* __restrict__ outp, long long sOz, int ldo, int M, int N,
              int K, float outScale) {
  __shared__ half_t As[64 * LDT];
  __shared__ half_t Bs[64 * LDT];
  const int z = blockIdx.z;
  A += (long long)(z / zdiv) * sAz1 + (long long)(z % zdiv) * sAz2;
  W += (long long)(z % zdiv) * sWz;
  const float* bz = bias + (long long)(z % zdiv) * sBz;
  const int m0 = blockIdx.x * 64, n0 = blockIdx.y * 64;
  const int tid = threadIdx.x, lane = tid & 31, wid = tid >> 5;
  const int lrow = tid >> 1, koff = (tid & 1) * 16;

  v8f acc[4] = {};

  for (int k0 = 0; k0 < K; k0 += 32) {
    __syncthreads();
    {  // stage A (f16, M-bounded)
      int gr = m0 + lrow;
      half_t* dst = &As[lrow * LDT + koff];
      if (gr < M) {
        stage16h(A + (long long)gr * lda + k0 + koff, dst);
      } else {
#pragma unroll
        for (int i = 0; i < 16; ++i) dst[i] = (half_t)0.0f;
      }
    }
    {  // stage W (fp32 -> f16); N multiple of 64
      const float* src = W + (long long)(n0 + lrow) * ldw + k0 + koff;
      half_t* dst = &Bs[lrow * LDT + koff];
#pragma unroll
      for (int i = 0; i < 16; ++i) dst[i] = (half_t)src[i];
      if (k0 + 32 < K) __builtin_prefetch(src + 32, 0, 1);
    }
    async_wait0();
    __syncthreads();
    v16h af = load_frag_a(As, LDT, wid * 16, 0, lane);
#pragma unroll
    for (int nb = 0; nb < 4; ++nb) {
      v16h bf = load_frag_b(Bs, LDT, nb * 16, 0, lane);
      acc[nb] = wmma_f16(af, bf, acc[nb]);
    }
  }

  const int rbase = m0 + wid * 16 + ((lane & 16) ? 8 : 0);
  const int cbase = n0 + (lane & 15);
#pragma unroll
  for (int nb = 0; nb < 4; ++nb) {
    int col = cbase + nb * 16;
    float bv = bz[col];
#pragma unroll
    for (int fr = 0; fr < 8; ++fr) {
      int row = rbase + fr;
      if (row < M) {
        float v = (acc[nb][fr] + bv) * outScale;
        outp[(long long)z * sOz + (long long)row * ldo + col] = (half_t)v;
      }
    }
  }
}

// ---------------------------------------------------------------------------
// Big-tile GEMM for MLP:  C[M,N] = A.W^T + bias ; EPI 1 = GELU->f16,
// EPI 2 = fp32 residual +=.  Block 256 thr / 8 waves, tile 128x128; each
// wave computes 32x64 (8 WMMA per K-panel from 6 fragment loads).
// ---------------------------------------------------------------------------
template <int EPI>
__global__ void __launch_bounds__(256)
gemm128_kernel(const half_t* __restrict__ A, int lda,
               const float* __restrict__ W, int ldw,
               const float* __restrict__ bias, void* __restrict__ outv,
               int ldo, int M, int N, int K) {
  __shared__ half_t As[128 * LDT];
  __shared__ half_t Ws[128 * LDT];
  const int m0 = blockIdx.x * 128, n0 = blockIdx.y * 128;
  const int tid = threadIdx.x, lane = tid & 31, wid = tid >> 5;
  const int mw = (wid & 3) * 32;   // wave M strip within tile
  const int nw = (wid >> 2) * 64;  // wave N strip within tile
  const int lrow = tid >> 1, koff = (tid & 1) * 16;

  v8f acc[2][4] = {};

  for (int k0 = 0; k0 < K; k0 += 32) {
    __syncthreads();
    {  // stage A 128x32 (f16, M-bounded)
      int gr = m0 + lrow;
      half_t* dst = &As[lrow * LDT + koff];
      if (gr < M) {
        stage16h(A + (long long)gr * lda + k0 + koff, dst);
      } else {
#pragma unroll
        for (int i = 0; i < 16; ++i) dst[i] = (half_t)0.0f;
      }
    }
    {  // stage W 128x32 (fp32 -> f16); N multiple of 128 in our uses
      const float* src = W + (long long)(n0 + lrow) * ldw + k0 + koff;
      half_t* dst = &Ws[lrow * LDT + koff];
#pragma unroll
      for (int i = 0; i < 16; ++i) dst[i] = (half_t)src[i];
      if (k0 + 32 < K) __builtin_prefetch(src + 32, 0, 1);
    }
    async_wait0();
    __syncthreads();
    v16h a0 = load_frag_a(As, LDT, mw, 0, lane);
    v16h a1 = load_frag_a(As, LDT, mw + 16, 0, lane);
#pragma unroll
    for (int nb = 0; nb < 4; ++nb) {
      v16h bf = load_frag_b(Ws, LDT, nw + nb * 16, 0, lane);
      acc[0][nb] = wmma_f16(a0, bf, acc[0][nb]);
      acc[1][nb] = wmma_f16(a1, bf, acc[1][nb]);
    }
  }

  const int cbase = n0 + nw + (lane & 15);
#pragma unroll
  for (int half = 0; half < 2; ++half) {
    const int rbase = m0 + mw + half * 16 + ((lane & 16) ? 8 : 0);
#pragma unroll
    for (int nb = 0; nb < 4; ++nb) {
      int col = cbase + nb * 16;
      float bv = bias[col];
#pragma unroll
      for (int fr = 0; fr < 8; ++fr) {
        int row = rbase + fr;
        if (row < M) {
          float v = acc[half][nb][fr] + bv;
          if (EPI == 1) {
            v = 0.5f * v * (1.0f + erff(v * 0.70710678118f));
            ((half_t*)outv)[(long long)row * ldo + col] = (half_t)v;
          } else {
            ((float*)outv)[(long long)row * ldo + col] += v;
          }
        }
      }
    }
  }
}

// ---------------------------------------------------------------------------
// Flash attention: grid (ceil(S/64), B*H), 128 thr / 4 waves.
// Q arrives pre-scaled by 1/8 (folded into Q projection). Online softmax;
// output += into residual (reference MSA has no output projection).
// ---------------------------------------------------------------------------
__global__ void __launch_bounds__(128)
flash_kernel(const half_t* __restrict__ Qg, const half_t* __restrict__ Kg,
             const half_t* __restrict__ Vg, float* __restrict__ resid) {
  __shared__ half_t Qs[64 * 72];
  __shared__ half_t Ks[64 * 72];
  __shared__ half_t Vs[64 * 72];  // transposed: Vs[e*72 + key]
  __shared__ half_t Ps[64 * 72];

  const int z = blockIdx.y;  // b*H + h
  const int qb0 = blockIdx.x * 64;
  const half_t* Qp = Qg + (long long)z * SS * DHEAD;
  const half_t* Kp = Kg + (long long)z * SS * DHEAD;
  const half_t* Vp = Vg + (long long)z * SS * DHEAD;
  const int tid = threadIdx.x, lane = tid & 31, wid = tid >> 5;
  const int lrow = tid >> 1, lcoff = (tid & 1) * 32;

  {  // stage Q tile (async)
    int qr = qb0 + lrow;
    half_t* dst = &Qs[lrow * 72 + lcoff];
    if (qr < SS) {
      stage32h(Qp + (long long)qr * DHEAD + lcoff, dst);
    } else {
#pragma unroll
      for (int i = 0; i < 32; ++i) dst[i] = (half_t)0.0f;
    }
  }

  v8f accO[4] = {};
  float mrow[8], lsum[8];
#pragma unroll
  for (int i = 0; i < 8; ++i) { mrow[i] = -1e30f; lsum[i] = 0.0f; }

  for (int kb0 = 0; kb0 < SS; kb0 += 64) {
    __syncthreads();
    {  // stage K tile (async) and transposed V tile (manual)
      int kr = kb0 + lrow;
      half_t* dk = &Ks[lrow * 72 + lcoff];
      if (kr < SS) {
        stage32h(Kp + (long long)kr * DHEAD + lcoff, dk);
        const half_t* sv = Vp + (long long)kr * DHEAD + lcoff;
#pragma unroll
        for (int i = 0; i < 32; ++i) Vs[(lcoff + i) * 72 + lrow] = sv[i];
      } else {
#pragma unroll
        for (int i = 0; i < 32; ++i) dk[i] = (half_t)0.0f;
#pragma unroll
        for (int i = 0; i < 32; ++i) Vs[(lcoff + i) * 72 + lrow] = (half_t)0.0f;
      }
    }
    async_wait0();
    __syncthreads();

    // S = Q . K^T
    v8f accS[4] = {};
#pragma unroll
    for (int k0 = 0; k0 < 64; k0 += 32) {
      v16h aq = load_frag_a(Qs, 72, wid * 16, k0, lane);
#pragma unroll
      for (int nb = 0; nb < 4; ++nb) {
        v16h bk = load_frag_b(Ks, 72, nb * 16, k0, lane);
        accS[nb] = wmma_f16(aq, bk, accS[nb]);
      }
    }

    // online softmax update (row = fr + (lane<16?0:8); 16-lane halves)
#pragma unroll
    for (int fr = 0; fr < 8; ++fr) {
      float sv[4];
      float rmax = -1e30f;
#pragma unroll
      for (int nb = 0; nb < 4; ++nb) {
        int key = kb0 + nb * 16 + (lane & 15);
        float s = accS[nb][fr];
        if (key >= SS) s = -1e30f;
        sv[nb] = s;
        rmax = fmaxf(rmax, s);
      }
      for (int m = 1; m < 16; m <<= 1) rmax = fmaxf(rmax, __shfl_xor(rmax, m, 32));
      float mnew = fmaxf(mrow[fr], rmax);
      float alpha = __expf(mrow[fr] - mnew);
      float psum = 0.0f;
      int prow = wid * 16 + fr + ((lane & 16) ? 8 : 0);
#pragma unroll
      for (int nb = 0; nb < 4; ++nb) {
        float p = __expf(sv[nb] - mnew);
        psum += p;
        Ps[prow * 72 + nb * 16 + (lane & 15)] = (half_t)p;
      }
      for (int m = 1; m < 16; m <<= 1) psum += __shfl_xor(psum, m, 32);
      lsum[fr] = lsum[fr] * alpha + psum;
      mrow[fr] = mnew;
#pragma unroll
      for (int nb = 0; nb < 4; ++nb) accO[nb][fr] *= alpha;
    }

    // O += P . V  (Ps strip is wave-private; same-wave LDS ops are in order)
#pragma unroll
    for (int k0 = 0; k0 < 64; k0 += 32) {
      v16h ap = load_frag_a(Ps, 72, wid * 16, k0, lane);
#pragma unroll
      for (int nb = 0; nb < 4; ++nb) {
        v16h bv = load_frag_b(Vs, 72, nb * 16, k0, lane);
        accO[nb] = wmma_f16(ap, bv, accO[nb]);
      }
    }
  }

  // epilogue: resid[b, q, h*64 + e] += O / lsum
  const int b = z / NH, h = z % NH;
  const int rrel = (lane & 16) ? 8 : 0;
#pragma unroll
  for (int nb = 0; nb < 4; ++nb) {
    int e = h * DHEAD + nb * 16 + (lane & 15);
#pragma unroll
    for (int fr = 0; fr < 8; ++fr) {
      int qr = qb0 + wid * 16 + rrel + fr;
      if (qr < SS) {
        float ov = accO[nb][fr] / lsum[fr];
        float* o = resid + ((long long)(b * SS + qr)) * DHID + e;
        *o += ov;
      }
    }
  }
}

// ---------------------------------------------------------------------------
// host launcher
// ---------------------------------------------------------------------------
extern "C" void kernel_launch(void* const* d_in, const int* in_sizes, int n_in,
                              void* d_out, int out_size, void* d_ws,
                              size_t ws_size, hipStream_t stream) {
  (void)in_sizes; (void)n_in; (void)out_size; (void)ws_size;
  const float* x    = (const float*)d_in[0];
  const float* ln1g = (const float*)d_in[1];
  const float* ln1b = (const float*)d_in[2];
  const float* qw   = (const float*)d_in[3];
  const float* qb   = (const float*)d_in[4];
  const float* kw   = (const float*)d_in[5];
  const float* kb   = (const float*)d_in[6];
  const float* vw   = (const float*)d_in[7];
  const float* vb   = (const float*)d_in[8];
  const float* ln2g = (const float*)d_in[9];
  const float* ln2b = (const float*)d_in[10];
  const float* w1   = (const float*)d_in[11];
  const float* b1   = (const float*)d_in[12];
  const float* w2   = (const float*)d_in[13];
  const float* b2   = (const float*)d_in[14];
  float* resid = (float*)d_out;

  char* ws = (char*)d_ws;
  size_t off = 0;
  auto take = [&](size_t bytes) {
    void* p = ws + off;
    off = (off + bytes + 255) & ~(size_t)255;
    return p;
  };
  const size_t actsz = (size_t)TT * DHID * sizeof(half_t);
  half_t* xn   = (half_t*)take(actsz);
  half_t* qbuf = (half_t*)take(actsz);
  half_t* kbuf = (half_t*)take(actsz);
  half_t* vbuf = (half_t*)take(actsz);
  half_t* hbuf = qbuf;  // MLP hidden reuses dead q/k/v region (contiguous)

  {  // residual = x
    int n4 = (TT * DHID) / 4;
    copy_f4_kernel<<<(n4 + 255) / 256, 256, 0, stream>>>((const float4*)x,
                                                         (float4*)resid, n4);
  }

  const int mtilesS = (SS + 63) / 64;  // 10
  const int CH = 4672;                 // MLP token chunk

  for (int l = 0; l < LNUM; ++l) {
    // ---- MSA ----
    ln_kernel<<<TT, 256, 0, stream>>>(resid, ln1g + l * DHID, ln1b + l * DHID, xn);

    dim3 gq(mtilesS, 1, BB * NH);
    const long long hW = (long long)DHEAD * DHEAD;
    gemm64_kernel<<<gq, 128, 0, stream>>>(
        xn, (long long)SS * DHID, (long long)DHEAD, NH, DHID,
        qw + (long long)l * NH * hW, hW, DHEAD, qb + (long long)l * NH * DHEAD,
        DHEAD, qbuf, (long long)SS * DHEAD, DHEAD, SS, DHEAD, DHEAD, 0.125f);
    gemm64_kernel<<<gq, 128, 0, stream>>>(
        xn, (long long)SS * DHID, (long long)DHEAD, NH, DHID,
        kw + (long long)l * NH * hW, hW, DHEAD, kb + (long long)l * NH * DHEAD,
        DHEAD, kbuf, (long long)SS * DHEAD, DHEAD, SS, DHEAD, DHEAD, 1.0f);
    gemm64_kernel<<<gq, 128, 0, stream>>>(
        xn, (long long)SS * DHID, (long long)DHEAD, NH, DHID,
        vw + (long long)l * NH * hW, hW, DHEAD, vb + (long long)l * NH * DHEAD,
        DHEAD, vbuf, (long long)SS * DHEAD, DHEAD, SS, DHEAD, DHEAD, 1.0f);

    dim3 gf(mtilesS, BB * NH);
    flash_kernel<<<gf, 128, 0, stream>>>(qbuf, kbuf, vbuf, resid);

    // ---- MLP ----
    ln_kernel<<<TT, 256, 0, stream>>>(resid, ln2g + l * DHID, ln2b + l * DHID, xn);

    for (int c0 = 0; c0 < TT; c0 += CH) {
      int mc = (TT - c0 < CH) ? (TT - c0) : CH;
      dim3 g1((mc + 127) / 128, MLPD / 128, 1);
      gemm128_kernel<1><<<g1, 256, 0, stream>>>(
          xn + (long long)c0 * DHID, DHID, w1 + (long long)l * MLPD * DHID,
          DHID, b1 + (long long)l * MLPD, hbuf, MLPD, mc, MLPD, DHID);
      dim3 g2((mc + 127) / 128, DHID / 128, 1);
      gemm128_kernel<2><<<g2, 256, 0, stream>>>(
          hbuf, MLPD, w2 + (long long)l * DHID * MLPD, MLPD,
          b2 + (long long)l * DHID, resid + (long long)c0 * DHID, DHID, mc,
          DHID, MLPD);
    }
  }
}
